// MyParticleNetwork_19636590477444
// MI455X (gfx1250) — compile-verified, matrix operand order checked by reading, not run on hardware
//
#include <hip/hip_runtime.h>

typedef __attribute__((ext_vector_type(16))) _Float16     v16h;
typedef __attribute__((ext_vector_type(8)))  float        v8f;
typedef __attribute__((ext_vector_type(4)))  unsigned int v4u;
typedef __attribute__((ext_vector_type(2)))  unsigned int v2u;

#define NF    6144
#define NB    4096
#define KN    96
#define TT    64
#define RADIUS 0.1125f
#define R2     (RADIUS * RADIUS)
#define BQR    0.1125f      /* reference compares squared dist to (unsquared) radius */
#define KLOC   245          /* int(6144 * 0.04) */
#define DTSTEP 0.02f
#define EPSG   1e-12f
#define CHUNK  1536

union HFrag { v16h v; v4u q[2]; };
union HPack { v4u q; _Float16 h[8]; };

__device__ __forceinline__ float sgnf(float x) {
    return (x > 0.f) ? 1.f : ((x < 0.f) ? -1.f : 0.f);
}

// ---------------------------------------------------------------------------
// location: count neighbors with d2 <= BQR (kth-NN > r  <=>  count < k)
// ---------------------------------------------------------------------------
__global__ void k_location(const float* __restrict__ pos, float* __restrict__ loc) {
    int i = blockIdx.x * blockDim.x + threadIdx.x;
    if (i >= NF) return;
    float px = pos[i * 3 + 0], py = pos[i * 3 + 1], pz = pos[i * 3 + 2];
    int cnt = 0;
    for (int j = 0; j < NF; ++j) {
        float dx = pos[j * 3 + 0] - px;
        float dy = pos[j * 3 + 1] - py;
        float dz = pos[j * 3 + 2] - pz;
        float d2 = dx * dx + dy * dy + dz * dz;
        cnt += (d2 <= BQR) ? 1 : 0;   // includes self, matching knn-with-self
    }
    loc[i] = (cnt < KLOC) ? 1.f : 0.f;
}

// ---------------------------------------------------------------------------
// integrate + build fluid feature rows (f32 stride 16, f16 stride 32 for d0 GEMM)
// ---------------------------------------------------------------------------
__global__ void k_integrate(const float* __restrict__ pos, const float* __restrict__ vel,
                            const float* __restrict__ loc, float* __restrict__ pos2,
                            float* __restrict__ ff32, _Float16* __restrict__ ff16) {
    int i = blockIdx.x * blockDim.x + threadIdx.x;
    if (i >= NF) return;
    float v0x = vel[i * 3 + 0], v0y = vel[i * 3 + 1], v0z = vel[i * 3 + 2];
    float v2x = v0x, v2y = v0y - DTSTEP * 9.81f, v2z = v0z;
    float p2x = pos[i * 3 + 0] + DTSTEP * (v2x + v0x) * 0.5f;
    float p2y = pos[i * 3 + 1] + DTSTEP * (v2y + v0y) * 0.5f;
    float p2z = pos[i * 3 + 2] + DTSTEP * (v2z + v0z) * 0.5f;
    pos2[i * 4 + 0] = p2x; pos2[i * 4 + 1] = p2y;
    pos2[i * 4 + 2] = p2z; pos2[i * 4 + 3] = 0.f;
    float f[16];
    #pragma unroll
    for (int c = 0; c < 16; ++c) f[c] = 0.f;
    f[0] = 1.f; f[1] = v2x; f[2] = v2y; f[3] = v2z; f[4] = loc[i];
    #pragma unroll
    for (int c = 0; c < 16; ++c) ff32[i * 16 + c] = f[c];
    #pragma unroll
    for (int c = 0; c < 32; ++c) ff16[i * 32 + c] = (c < 16) ? (_Float16)f[c] : (_Float16)0.f;
}

// ---------------------------------------------------------------------------
// deterministic radius-neighbor scan (first up to 96 in index order; -1 pad)
// ---------------------------------------------------------------------------
__global__ void k_neighbors(const float* __restrict__ q4, const float* __restrict__ pts,
                            int pstride, int npts, int excludeSelf, int* __restrict__ idxOut) {
    int i = blockIdx.x * blockDim.x + threadIdx.x;
    if (i >= NF) return;
    float qx = q4[i * 4 + 0], qy = q4[i * 4 + 1], qz = q4[i * 4 + 2];
    int cnt = 0;
    for (int j = 0; j < npts; ++j) {
        if (excludeSelf && j == i) continue;
        float dx = pts[j * pstride + 0] - qx;
        float dy = pts[j * pstride + 1] - qy;
        float dz = pts[j * pstride + 2] - qz;
        float d2 = dx * dx + dy * dy + dz * dz;
        if (d2 <= R2 && cnt < KN) { idxOut[i * KN + cnt] = j; ++cnt; }
    }
    for (; cnt < KN; ++cnt) idxOut[i * KN + cnt] = -1;
}

// ---------------------------------------------------------------------------
// per-(query,neighbor) geometry: poly6 window, ball->cube, trilinear taps
// ---------------------------------------------------------------------------
__global__ void k_geom(const float* __restrict__ q4, const float* __restrict__ pts,
                       int pstride, const int* __restrict__ idx,
                       unsigned char* __restrict__ taps, _Float16* __restrict__ alpha) {
    int gid = blockIdx.x * blockDim.x + threadIdx.x;
    if (gid >= NF * KN) return;
    int i = gid / KN;
    int nb = idx[gid];
    unsigned char tp[8];
    float al[8];
    if (nb < 0) {
        #pragma unroll
        for (int j = 0; j < 8; ++j) { tp[j] = 0; al[j] = 0.f; }
    } else {
        float x = (pts[nb * pstride + 0] - q4[i * 4 + 0]) * (1.f / RADIUS);
        float y = (pts[nb * pstride + 1] - q4[i * 4 + 1]) * (1.f / RADIUS);
        float z = (pts[nb * pstride + 2] - q4[i * 4 + 2]) * (1.f / RADIUS);
        float r2 = x * x + y * y + z * z;
        float w = 1.f - r2; w = w * w * w;
        float window = fminf(fmaxf(w, 0.f), 1.f);
        float norm = sqrtf(r2 + EPSG);
        float sq_xy = x * x + y * y;
        bool top = (1.25f * z * z) > sq_xy;
        float s_top = sqrtf(3.f * norm / (norm + fabsf(z) + EPSG));
        float s_side = norm / sqrtf(sq_xy + EPSG);
        float s = top ? s_top : s_side;
        float cx = x * s, cy = y * s;
        float cz = top ? sgnf(z) * norm : 1.5f * z;
        if (r2 < EPSG) { cx = 0.f; cy = 0.f; cz = 0.f; }
        float n_xy = sqrtf(cx * cx + cy * cy + EPSG);
        bool xdom = fabsf(cy) <= fabsf(cx);
        float safe_cx = (fabsf(cx) > EPSG) ? cx : 1.f;
        float safe_cy = (fabsf(cy) > EPSG) ? cy : 1.f;
        const float c4pi = 1.27323954473516f; // 4/pi
        float ux = sgnf(cx) * n_xy;
        float vx = ux * c4pi * atanf(cy / safe_cx);
        float vy = sgnf(cy) * n_xy;
        float uy = vy * c4pi * atanf(cx / safe_cy);
        float u = xdom ? ux : uy;
        float v = xdom ? vx : vy;
        if (cx * cx + cy * cy < EPSG) { u = 0.f; v = 0.f; }
        float cc[3] = { (u + 1.f) * 1.5f, (v + 1.f) * 1.5f, (cz + 1.f) * 1.5f };
        int   i0[3]; float fr[3];
        #pragma unroll
        for (int d = 0; d < 3; ++d) {
            float c = fminf(fmaxf(cc[d], 0.f), 3.f);
            int ii = (int)floorf(c);
            ii = ii < 0 ? 0 : (ii > 2 ? 2 : ii);
            i0[d] = ii; fr[d] = c - (float)ii;
        }
        #pragma unroll
        for (int cr = 0; cr < 8; ++cr) {
            int ci = (cr >> 2) & 1, cj = (cr >> 1) & 1, ck = cr & 1;
            tp[cr] = (unsigned char)(((i0[0] + ci) * 4 + (i0[1] + cj)) * 4 + (i0[2] + ck));
            float cw = (ci ? fr[0] : 1.f - fr[0]) * (cj ? fr[1] : 1.f - fr[1]) *
                       (ck ? fr[2] : 1.f - fr[2]);
            al[cr] = cw * window;
        }
    }
    #pragma unroll
    for (int j = 0; j < 8; ++j) {
        taps[gid * 8 + j]  = tp[j];
        alpha[gid * 8 + j] = (_Float16)al[j];
    }
}

// ---------------------------------------------------------------------------
// repack weights: B[Kpad][64] f16; rows [0,Tcin): Wc[t][c][d], then stacked Wdense
// ---------------------------------------------------------------------------
__global__ void k_prepB(const float* __restrict__ Wc, const float* __restrict__ Wdn,
                        int Cin, int Cinp, int Cout, int Tcin, int KdimPad,
                        _Float16* __restrict__ B) {
    int gid = blockIdx.x * blockDim.x + threadIdx.x;
    if (gid >= KdimPad * 64) return;
    int r = gid >> 6, d = gid & 63;
    float v = 0.f;
    if (r < Tcin) {
        int t = r / Cinp, c = r - t * Cinp;
        if (Wc && c < Cin && d < Cout) v = Wc[(t * Cin + c) * Cout + d];
    } else {
        int c = r - Tcin;
        if (Wdn && c < Cin && d < Cout) v = Wdn[c * Cout + d];
    }
    B[gid] = (_Float16)v;
}

// ---------------------------------------------------------------------------
// Stage B (CINP templated so n-tile count is constexpr -> accumulators stay in
// VGPRs, loops fully unroll; round-2 runtime version spilled acc[] to scratch).
// Per-query  A[t,c] = sum_k Wd[k][t] * F[k][c]  via WMMA (64 x CINP x 96).
// Wd / F built K-contiguous ([t][k], [c][k]): each fragment = 2x ds_load_b128.
// ---------------------------------------------------------------------------
template <int CINP>
__global__ __launch_bounds__(128) void k_stageB(
    const int* __restrict__ idx, const unsigned char* __restrict__ taps,
    const _Float16* __restrict__ alpha, const float* __restrict__ feats, int fstride,
    int Cin, int useRelu, int appendSelf,
    _Float16* __restrict__ Aout, int KdimPad, int q0) {
    constexpr int NTILES = CINP / 16;
    __shared__ __attribute__((aligned(16))) _Float16 Wd2[TT * KN];     // [t][k]
    __shared__ __attribute__((aligned(16))) _Float16 Fs2[CINP * KN];   // [c][k]
    __shared__ __attribute__((aligned(16))) _Float16 St [TT * CINP];   // staging [t][c]
    int q   = q0 + blockIdx.x;
    int tid = threadIdx.x;

    {   // zero Wd2 (v4u = 8 halves per store)
        const v4u z4 = {0u, 0u, 0u, 0u};
        v4u* wz = (v4u*)Wd2;
        for (int i = tid; i < TT * KN / 8; i += 128) wz[i] = z4;
    }
    __syncthreads();
    if (tid < KN) {                        // (t,k)-unique scatter: no conflicts
        int base = (q * KN + tid) * 8;
        v2u tpq = *(const v2u*)(taps + base);
        HPack al; al.q = *(const v4u*)(alpha + base);
        #pragma unroll
        for (int j = 0; j < 8; ++j) {
            unsigned int word = (j < 4) ? tpq.x : tpq.y;
            int tp = (int)((word >> (8 * (j & 3))) & 0xffu);
            Wd2[tp * KN + tid] = al.h[j];
        }
    }
    // gather neighbor features, transposed to [c][k]
    for (int e = tid; e < KN * CINP; e += 128) {
        int k = e / CINP, c = e - k * CINP;
        int nb = idx[q * KN + k];
        float v = 0.f;
        if (nb >= 0 && c < Cin) {
            v = feats[nb * fstride + c];
            if (useRelu) v = fmaxf(v, 0.f);
        }
        Fs2[c * KN + k] = (_Float16)v;
    }
    __syncthreads();

    int wave = tid >> 5, lane = tid & 31;
    int hoff = (lane >= 16) ? 8 : 0;
    int mrow = lane & 15;
    int t0   = wave * 16;

    const v8f vzero = {0.f, 0.f, 0.f, 0.f, 0.f, 0.f, 0.f, 0.f};
    v8f acc[NTILES];
    #pragma unroll
    for (int i = 0; i < NTILES; ++i) acc[i] = vzero;

    #pragma unroll
    for (int kb = 0; kb < KN; kb += 32) {
        HFrag af;
        const _Float16* ap = &Wd2[(t0 + mrow) * KN + kb + hoff];
        af.q[0] = *(const v4u*)(ap);
        af.q[1] = *(const v4u*)(ap + 16);
        #pragma unroll
        for (int nt = 0; nt < NTILES; ++nt) {
            HFrag bf;
            const _Float16* bp = &Fs2[(nt * 16 + mrow) * KN + kb + hoff];
            bf.q[0] = *(const v4u*)(bp);
            bf.q[1] = *(const v4u*)(bp + 16);
            acc[nt] = __builtin_amdgcn_wmma_f32_16x16x32_f16(
                false, af.v, false, bf.v, (short)0, acc[nt], false, false);
        }
    }
    // stage results in LDS [t][c]
    #pragma unroll
    for (int nt = 0; nt < NTILES; ++nt) {
        #pragma unroll
        for (int r = 0; r < 8; ++r) {
            int t = t0 + r + hoff;
            int c = nt * 16 + mrow;
            St[t * CINP + c] = (_Float16)acc[nt][r];
        }
    }
    __syncthreads();
    // coalesced b128 flush of the A-row
    size_t rowBase = (size_t)(q - q0) * (size_t)KdimPad;
    constexpr int NVEC = (TT * CINP) / 8;  // multiples of 8 halves
    for (int u = tid; u < NVEC; u += 128) {
        *(v4u*)(Aout + rowBase + (size_t)u * 8) = *(const v4u*)(&St[u * 8]);
    }
    // appended self-features / padding columns
    for (int col = TT * CINP + tid; col < KdimPad; col += 128) {
        int c = col - TT * CINP;
        float v = 0.f;
        if (appendSelf && c < Cin) {
            v = feats[q * fstride + c];
            if (useRelu) v = fmaxf(v, 0.f);
        }
        Aout[rowBase + col] = (_Float16)v;
    }
}

// ---------------------------------------------------------------------------
// Stage C: tiled WMMA GEMM  out[64 x 64] = A[64 x K] * B[K x 64]
// A tile kept [m][k]; B tile transposed into [n][k] at LDS-store time so both
// fragments load as 2x ds_load_b128. Fused bias / residual / column offset.
// ---------------------------------------------------------------------------
__global__ __launch_bounds__(128) void k_gemm(
    const _Float16* __restrict__ A, int lda, const _Float16* __restrict__ B, int Kdim,
    const float* __restrict__ bias, const float* __restrict__ res, int resStride,
    float* __restrict__ out, int outStride, int colOff, int Cout) {
    __shared__ __attribute__((aligned(16))) _Float16 At[64 * 32];   // [m][k]
    __shared__ __attribute__((aligned(16))) _Float16 Bt[64 * 32];   // [n][k]
    int blk = blockIdx.x, tid = threadIdx.x;
    int wave = tid >> 5, lane = tid & 31;
    int hoff = (lane >= 16) ? 8 : 0;
    int mrow = lane & 15;
    int t0   = wave * 16;
    const v8f vzero = {0.f, 0.f, 0.f, 0.f, 0.f, 0.f, 0.f, 0.f};
    v8f acc[4];
    #pragma unroll
    for (int i = 0; i < 4; ++i) acc[i] = vzero;

    for (int kb = 0; kb < Kdim; kb += 32) {
        if (kb + 32 < Kdim) {   // prefetch next K tile (global_prefetch_b8)
            __builtin_prefetch(A + (size_t)(blk * 64 + (tid & 63)) * lda + kb + 32, 0, 1);
            __builtin_prefetch(B + (size_t)(kb + 32 + (tid >> 2)) * 64, 0, 1);
        }
        __syncthreads();
        // A tile: contiguous dword copies (16 dwords per row, 64 rows)
        unsigned int* au = (unsigned int*)At;
        for (int u = tid; u < 1024; u += 128) {
            int m  = u >> 4;
            int kp = (u & 15) << 1;
            au[u] = *(const unsigned int*)(A + (size_t)(blk * 64 + m) * lda + kb + kp);
        }
        // B tile: load [k][n] dwords, store transposed halves into [n][k]
        for (int u = tid; u < 1024; u += 128) {
            int k  = u >> 5;            // 0..31
            int np = (u & 31) << 1;     // 0..62
            HPack cv; cv.q.x = *(const unsigned int*)(B + (size_t)(kb + k) * 64 + np);
            cv.q.y = cv.q.z = cv.q.w = 0u;
            Bt[np * 32 + k]       = cv.h[0];
            Bt[(np + 1) * 32 + k] = cv.h[1];
        }
        __syncthreads();
        HFrag af;
        const _Float16* ap = &At[(t0 + mrow) * 32 + hoff];
        af.q[0] = *(const v4u*)(ap);
        af.q[1] = *(const v4u*)(ap + 16);
        #pragma unroll
        for (int nt = 0; nt < 4; ++nt) {
            HFrag bf;
            const _Float16* bp = &Bt[(nt * 16 + mrow) * 32 + hoff];
            bf.q[0] = *(const v4u*)(bp);
            bf.q[1] = *(const v4u*)(bp + 16);
            acc[nt] = __builtin_amdgcn_wmma_f32_16x16x32_f16(
                false, af.v, false, bf.v, (short)0, acc[nt], false, false);
        }
    }
    #pragma unroll
    for (int nt = 0; nt < 4; ++nt) {
        #pragma unroll
        for (int r = 0; r < 8; ++r) {
            int m = blk * 64 + t0 + r + hoff;
            int n = nt * 16 + mrow;
            if (n < Cout) {
                float vv = acc[nt][r];
                if (bias) vv += bias[n];
                if (res)  vv += res[(size_t)m * resStride + n];
                out[(size_t)m * outStride + colOff + n] = vv;
            }
        }
    }
}

// ---------------------------------------------------------------------------
// finalize: pos_out = pos2 + x3/128 ; vel_out = (pos_out - pos)/DT
// ---------------------------------------------------------------------------
__global__ void k_final(const float* __restrict__ pos, const float* __restrict__ pos2,
                        const float* __restrict__ x3, float* __restrict__ out) {
    int i = blockIdx.x * blockDim.x + threadIdx.x;
    if (i >= NF) return;
    #pragma unroll
    for (int d = 0; d < 3; ++d) {
        float po = pos2[i * 4 + d] + x3[i * 16 + d] * (1.f / 128.f);
        out[i * 3 + d]          = po;
        out[NF * 3 + i * 3 + d] = (po - pos[i * 3 + d]) * (1.f / DTSTEP);
    }
}

// ---------------------------------------------------------------------------
extern "C" void kernel_launch(void* const* d_in, const int* in_sizes, int n_in,
                              void* d_out, int out_size, void* d_ws, size_t ws_size,
                              hipStream_t stream) {
    (void)in_sizes; (void)n_in; (void)out_size; (void)ws_size;
    const float* pos       = (const float*)d_in[0];
    const float* vel       = (const float*)d_in[1];
    const float* box       = (const float*)d_in[2];
    const float* box_feats = (const float*)d_in[3];
    const float* Wc0f = (const float*)d_in[4];
    const float* Wc0o = (const float*)d_in[5];
    const float* Wc1  = (const float*)d_in[6];
    const float* Wc2  = (const float*)d_in[7];
    const float* Wc3  = (const float*)d_in[8];
    const float* Wd0  = (const float*)d_in[9];
    const float* bd0  = (const float*)d_in[10];
    const float* Wd1  = (const float*)d_in[11];
    const float* bd1  = (const float*)d_in[12];
    const float* Wd2  = (const float*)d_in[13];
    const float* bd2  = (const float*)d_in[14];
    const float* Wd3  = (const float*)d_in[15];
    const float* bd3  = (const float*)d_in[16];

    char* w = (char*)d_ws;
    size_t off = 0;
    auto alloc = [&](size_t bytes) -> char* {
        char* p = w + off;
        off = (off + bytes + 255) & ~(size_t)255;
        return p;
    };
    float*         loc      = (float*)alloc((size_t)NF * 4);
    float*         pos2     = (float*)alloc((size_t)NF * 4 * 4);
    float*         ff32     = (float*)alloc((size_t)NF * 16 * 4);
    _Float16*      ff16     = (_Float16*)alloc((size_t)NF * 32 * 2);
    int*           idx_ff   = (int*)alloc((size_t)NF * KN * 4);
    int*           idx_bf   = (int*)alloc((size_t)NF * KN * 4);
    unsigned char* taps_ff  = (unsigned char*)alloc((size_t)NF * KN * 8);
    unsigned char* taps_bf  = (unsigned char*)alloc((size_t)NF * KN * 8);
    _Float16*      alpha_ff = (_Float16*)alloc((size_t)NF * KN * 8 * 2);
    _Float16*      alpha_bf = (_Float16*)alloc((size_t)NF * KN * 8 * 2);
    _Float16*      B0o      = (_Float16*)alloc((size_t)1024 * 64 * 2);
    _Float16*      B0f      = (_Float16*)alloc((size_t)1024 * 64 * 2);
    _Float16*      Bd0      = (_Float16*)alloc((size_t)32 * 64 * 2);
    _Float16*      B1       = (_Float16*)alloc((size_t)6240 * 64 * 2);
    _Float16*      B2       = (_Float16*)alloc((size_t)4160 * 64 * 2);
    _Float16*      B3       = (_Float16*)alloc((size_t)4160 * 64 * 2);
    float*         x0       = (float*)alloc((size_t)NF * 96 * 4);
    float*         x1       = (float*)alloc((size_t)NF * 64 * 4);
    float*         x2       = (float*)alloc((size_t)NF * 64 * 4);
    float*         x3       = (float*)alloc((size_t)NF * 16 * 4);
    _Float16*      Abuf     = (_Float16*)alloc((size_t)CHUNK * 6240 * 2);

    k_location<<<NF / 256, 256, 0, stream>>>(pos, loc);
    k_integrate<<<NF / 256, 256, 0, stream>>>(pos, vel, loc, pos2, ff32, ff16);
    k_neighbors<<<NF / 256, 256, 0, stream>>>(pos2, pos2, 4, NF, 1, idx_ff);
    k_neighbors<<<NF / 256, 256, 0, stream>>>(pos2, box, 3, NB, 0, idx_bf);
    k_geom<<<(NF * KN + 255) / 256, 256, 0, stream>>>(pos2, pos2, 4, idx_ff, taps_ff, alpha_ff);
    k_geom<<<(NF * KN + 255) / 256, 256, 0, stream>>>(pos2, box, 3, idx_bf, taps_bf, alpha_bf);

    k_prepB<<<(1024 * 64 + 255) / 256, 256, 0, stream>>>(Wc0o, nullptr, 3, 16, 32, 1024, 1024, B0o);
    k_prepB<<<(1024 * 64 + 255) / 256, 256, 0, stream>>>(Wc0f, nullptr, 5, 16, 32, 1024, 1024, B0f);
    k_prepB<<<(32 * 64 + 255) / 256, 256, 0, stream>>>(nullptr, Wd0, 5, 16, 32, 0, 32, Bd0);
    k_prepB<<<(6240 * 64 + 255) / 256, 256, 0, stream>>>(Wc1, Wd1, 96, 96, 64, 6144, 6240, B1);
    k_prepB<<<(4160 * 64 + 255) / 256, 256, 0, stream>>>(Wc2, Wd2, 64, 64, 64, 4096, 4160, B2);
    k_prepB<<<(4160 * 64 + 255) / 256, 256, 0, stream>>>(Wc3, Wd3, 64, 64, 3, 4096, 4160, B3);

    // d0 = fluid_feats @ Wd0 + bd0  -> x0[:, 64:96]
    k_gemm<<<NF / 64, 128, 0, stream>>>(ff16, 32, Bd0, 32, bd0, nullptr, 0, x0, 96, 64, 32);

    // layer 0 convs -> x0[:,0:32] (box) and x0[:,32:64] (fluid)
    for (int c0 = 0; c0 < NF; c0 += CHUNK) {
        k_stageB<16><<<CHUNK, 128, 0, stream>>>(idx_bf, taps_bf, alpha_bf, box_feats, 3, 3,
                                                0, 0, Abuf, 1024, c0);
        k_gemm<<<CHUNK / 64, 128, 0, stream>>>(Abuf, 1024, B0o, 1024, nullptr, nullptr, 0,
                                               x0 + (size_t)c0 * 96, 96, 0, 32);
        k_stageB<16><<<CHUNK, 128, 0, stream>>>(idx_ff, taps_ff, alpha_ff, ff32, 16, 5,
                                                0, 0, Abuf, 1024, c0);
        k_gemm<<<CHUNK / 64, 128, 0, stream>>>(Abuf, 1024, B0f, 1024, nullptr, nullptr, 0,
                                               x0 + (size_t)c0 * 96, 96, 32, 32);
    }
    // layer 1: x1 = cconv(relu x0) + relu(x0)@Wd1 + bd1
    for (int c0 = 0; c0 < NF; c0 += CHUNK) {
        k_stageB<96><<<CHUNK, 128, 0, stream>>>(idx_ff, taps_ff, alpha_ff, x0, 96, 96,
                                                1, 1, Abuf, 6240, c0);
        k_gemm<<<CHUNK / 64, 128, 0, stream>>>(Abuf, 6240, B1, 6240, bd1, nullptr, 0,
                                               x1 + (size_t)c0 * 64, 64, 0, 64);
    }
    // layer 2: x2 = cconv(relu x1) + relu(x1)@Wd2 + bd2 + x1 (residual)
    for (int c0 = 0; c0 < NF; c0 += CHUNK) {
        k_stageB<64><<<CHUNK, 128, 0, stream>>>(idx_ff, taps_ff, alpha_ff, x1, 64, 64,
                                                1, 1, Abuf, 4160, c0);
        k_gemm<<<CHUNK / 64, 128, 0, stream>>>(Abuf, 4160, B2, 4160, bd2,
                                               x1 + (size_t)c0 * 64, 64,
                                               x2 + (size_t)c0 * 64, 64, 0, 64);
    }
    // layer 3: x3 = cconv(relu x2) + relu(x2)@Wd3 + bd3  (Cout = 3)
    for (int c0 = 0; c0 < NF; c0 += CHUNK) {
        k_stageB<64><<<CHUNK, 128, 0, stream>>>(idx_ff, taps_ff, alpha_ff, x2, 64, 64,
                                                1, 1, Abuf, 4160, c0);
        k_gemm<<<CHUNK / 64, 128, 0, stream>>>(Abuf, 4160, B3, 4160, bd3, nullptr, 0,
                                               x3 + (size_t)c0 * 16, 16, 0, 3);
    }

    k_final<<<NF / 256, 256, 0, stream>>>(pos, pos2, x3, (float*)d_out);
}
